// CasualSelfAttention_16681652978116
// MI455X (gfx1250) — compile-verified, matrix operand order checked by reading, not run on hardware
//
#include <hip/hip_runtime.h>

// ---------- types ----------
typedef __bf16 bf16;
typedef __attribute__((ext_vector_type(16))) __bf16 v16bf;
typedef __attribute__((ext_vector_type(8)))  __bf16 bf16x8;   // 16 bytes
typedef __attribute__((ext_vector_type(4)))  __bf16 bf16x4;   // 8 bytes
typedef __attribute__((ext_vector_type(8)))  float  v8f;

union Frag16 { v16bf v; bf16x8 h[2]; };

__device__ __forceinline__ v8f wmma_bf16(const Frag16& a, const Frag16& b, v8f c) {
  // D(f32 16x16) = A(bf16 16x32) * B(bf16 32x16) + C
  return __builtin_amdgcn_wmma_f32_16x16x32_bf16(false, a.v, false, b.v,
                                                 (short)0, c, false, false);
}

// ---------- problem constants ----------
#define BQ   4
#define SEQ  2048
#define CDIM 1024
#define NH   16
#define DH   64
#define N3C  3072

// ---------- Tensor Data Mover (TDM) support ----------
#if __has_builtin(__builtin_amdgcn_tensor_load_to_lds) && \
    __has_builtin(__builtin_amdgcn_s_wait_tensorcnt)
#define USE_TDM 1
#else
#define USE_TDM 0
#endif

#if USE_TDM
typedef unsigned int v4ui __attribute__((ext_vector_type(4)));
typedef int          v8i  __attribute__((ext_vector_type(8)));
typedef int          v4i  __attribute__((ext_vector_type(4)));

// 2D bf16 tile DMA: global (row_stride elems between rows) -> LDS (contiguous
// rows + D# padding: pad_amount DWORDs appended every pad_interval DWORDs;
// codes per CDNA5 D# group1 encoding).
__device__ __forceinline__ void tdm_load_2d_bf16(
    unsigned int lds_off, const void* gptr,
    unsigned int tile_w, unsigned int tile_h,
    unsigned int tensor_w, unsigned int tensor_h,
    unsigned long long row_stride_elems,
    unsigned int pad_interval_code, unsigned int pad_amount_code)
{
  unsigned long long ga = (unsigned long long)(uintptr_t)gptr;
  // D# group 0: count=1 | lds_addr | global_addr[56:0] | type=2 ("image")
  v4ui g0 = { 1u,
              lds_off,
              (unsigned int)ga,
              (unsigned int)((ga >> 32) & 0x1FFFFFFu) | (2u << 30) };
  // D# group 1
  unsigned int w0 = (1u << 16)                       // data_size = 2 bytes
                  | (1u << 20)                       // pad_enable
                  | (pad_interval_code << 22)
                  | (pad_amount_code << 25);
  v8i g1 = { (int)w0,
             (int)((tensor_w & 0xFFFFu) << 16),                          // dim0[15:0]
             (int)(((tensor_w >> 16) & 0xFFFFu) | ((tensor_h & 0xFFFFu) << 16)),
             (int)(((tensor_h >> 16) & 0xFFFFu) | ((tile_w & 0xFFFFu) << 16)),
             (int)(tile_h & 0xFFFFu),                                    // tile_dim1 (tile_dim2=0)
             (int)(unsigned int)(row_stride_elems & 0xFFFFFFFFull),      // dim0_stride[31:0]
             (int)(unsigned int)((row_stride_elems >> 32) & 0xFFFFull),  // dim0_stride[47:32]
             0 };
  v4i gz = { 0, 0, 0, 0 };
#if defined(__clang_major__) && (__clang_major__ >= 23)
  v8i gz8 = { 0, 0, 0, 0, 0, 0, 0, 0 };
  __builtin_amdgcn_tensor_load_to_lds(g0, g1, gz, gz, gz8, 0);
#else
  __builtin_amdgcn_tensor_load_to_lds(g0, g1, gz, gz, 0);
#endif
}
#endif  // USE_TDM

// =====================================================================
// Causal flash attention. 1 WG = 64 query rows of one (b,h); 4 waves,
// each owning 16 rows. K/V tiles DMA'd by the TDM into ping-pong LDS
// buffers: block i+1 streams in (TENSORcnt) while block i runs WMMA +
// online softmax.
// =====================================================================
__global__ __launch_bounds__(128) void flash_attn_kernel(
    const bf16* __restrict__ qb, const bf16* __restrict__ kb,
    const bf16* __restrict__ vb, bf16* __restrict__ att)
{
  constexpr int SP = 72;                          // padded LDS row stride
  __shared__ alignas(16) bf16 Ks[2][64 * SP];     // K tile [s][d], ping-pong
  __shared__ alignas(16) bf16 Vs[2][64 * SP];     // V^T tile [d][s], ping-pong
  __shared__ alignas(16) bf16 Ps[4][16 * SP];     // per-wave P bounce [m][s]
  const int bh   = blockIdx.x;                    // b*NH + h
  const int qblk = blockIdx.y;
  const int tid = threadIdx.x, wave = tid >> 5, lane = tid & 31;
  const int lr = lane & 15, hi = lane >> 4;
  const int kc = hi * 8;
  const size_t base = (size_t)bh * SEQ * DH;
  const int q0 = qblk * 64;
  const bf16* kg = kb + base;                     // K rows [SEQ][DH]
  const bf16* vg = vb + (size_t)bh * DH * SEQ;    // V^T rows [DH][SEQ]

  // Q fragments (2x 16x32 over D=64), kept in registers for all key blocks
  Frag16 qa[2];
  {
    const bf16* qrow = qb + base + (size_t)(q0 + wave * 16 + lr) * DH;
    qa[0].h[0] = *(const bf16x8*)&qrow[kc];
    qa[0].h[1] = *(const bf16x8*)&qrow[kc + 16];
    qa[1].h[0] = *(const bf16x8*)&qrow[32 + kc];
    qa[1].h[1] = *(const bf16x8*)&qrow[32 + kc + 16];
  }

  const v8f zero = {0.f,0.f,0.f,0.f,0.f,0.f,0.f,0.f};
  v8f o[4];
  for (int j = 0; j < 4; ++j) o[j] = zero;
  float mrow[8], lsum[8];
  for (int r = 0; r < 8; ++r) { mrow[r] = -3.0e38f; lsum[r] = 0.f; }

#if USE_TDM
  // prologue: DMA first K/V block into buffer 0
  // (pad 4 DWORDs every 32 DWORDs: row stride 64 -> 72 elems)
  if (wave == 0) {
    tdm_load_2d_bf16((unsigned int)(uintptr_t)&Ks[0][0], kg,
                     DH, 64, DH, SEQ, DH, 4u, 3u);
    tdm_load_2d_bf16((unsigned int)(uintptr_t)&Vs[0][0], vg,
                     64, DH, SEQ, DH, SEQ, 4u, 3u);
  }
#endif

  for (int s0 = 0; s0 <= q0; s0 += 64) {
    const int ib = (s0 >> 6) & 1;
#if USE_TDM
    if (wave == 0) __builtin_amdgcn_s_wait_tensorcnt(0);  // buf[ib] landed
    __syncthreads();   // all waves: see buf[ib]; done reading buf[ib^1]
    if (wave == 0 && s0 + 64 <= q0) {                     // stream next block
      const int nb = ib ^ 1;
      tdm_load_2d_bf16((unsigned int)(uintptr_t)&Ks[nb][0],
                       kg + (size_t)(s0 + 64) * DH,
                       DH, 64, DH, SEQ, DH, 4u, 3u);
      tdm_load_2d_bf16((unsigned int)(uintptr_t)&Vs[nb][0],
                       vg + (s0 + 64),
                       64, DH, SEQ, DH, SEQ, 4u, 3u);
    }
#else
    __syncthreads();
#pragma unroll
    for (int i = 0; i < 4; ++i) {
      int srow = tid >> 1;
      int p = ((tid & 1) * 4 + i) * 8;
      *(bf16x8*)&Ks[ib][srow * SP + p] =
          *(const bf16x8*)&kg[(size_t)(s0 + srow) * DH + p];
      *(bf16x8*)&Vs[ib][srow * SP + p] =
          *(const bf16x8*)&vg[(size_t)srow * SEQ + s0 + p];
    }
    __syncthreads();
#endif
    const bf16* Kc = &Ks[ib][0];
    const bf16* Vc = &Vs[ib][0];

    // S = Q @ K^T : 16x64 per wave (4 n-tiles x 2 k-steps)
    v8f sacc[4];
    for (int g = 0; g < 4; ++g) sacc[g] = zero;
#pragma unroll
    for (int g = 0; g < 4; ++g) {
      Frag16 b0, b1;
      int nr = g * 16 + lr;
      b0.h[0] = *(const bf16x8*)&Kc[nr * SP + kc];
      b0.h[1] = *(const bf16x8*)&Kc[nr * SP + kc + 16];
      b1.h[0] = *(const bf16x8*)&Kc[nr * SP + 32 + kc];
      b1.h[1] = *(const bf16x8*)&Kc[nr * SP + 32 + kc + 16];
      sacc[g] = wmma_bf16(qa[0], b0, sacc[g]);
      sacc[g] = wmma_bf16(qa[1], b1, sacc[g]);
    }

    if (s0 == q0) {   // diagonal block: causal mask
#pragma unroll
      for (int g = 0; g < 4; ++g)
#pragma unroll
        for (int r = 0; r < 8; ++r) {
          int ml = wave * 16 + r + hi * 8;
          int nl = g * 16 + lr;
          if (nl > ml) sacc[g][r] = -3.0e38f;
        }
    }

    // online softmax (row stats across 16 lanes of the half-wave)
    float alpha[8];
#pragma unroll
    for (int r = 0; r < 8; ++r) {
      float mx = fmaxf(fmaxf(sacc[0][r], sacc[1][r]),
                       fmaxf(sacc[2][r], sacc[3][r]));
#pragma unroll
      for (int off = 1; off < 16; off <<= 1)
        mx = fmaxf(mx, __shfl_xor(mx, off, 16));
      float nm = fmaxf(mrow[r], mx);
      alpha[r] = __expf(mrow[r] - nm);
      mrow[r] = nm;
    }
#pragma unroll
    for (int r = 0; r < 8; ++r) {
      float rs = 0.f;
#pragma unroll
      for (int g = 0; g < 4; ++g) {
        float p = __expf(sacc[g][r] - mrow[r]);
        sacc[g][r] = p;
        rs += p;
      }
#pragma unroll
      for (int off = 1; off < 16; off <<= 1) rs += __shfl_xor(rs, off, 16);
      lsum[r] = lsum[r] * alpha[r] + rs;
    }
#pragma unroll
    for (int j = 0; j < 4; ++j)
#pragma unroll
      for (int r = 0; r < 8; ++r) o[j][r] *= alpha[r];

    // P (D-layout f32) -> wave-private LDS -> A-layout bf16 fragments
#pragma unroll
    for (int g = 0; g < 4; ++g)
#pragma unroll
      for (int r = 0; r < 8; ++r)
        Ps[wave][(r + hi * 8) * SP + g * 16 + lr] = (bf16)sacc[g][r];
    Frag16 pa[2];
    pa[0].h[0] = *(const bf16x8*)&Ps[wave][lr * SP + kc];
    pa[0].h[1] = *(const bf16x8*)&Ps[wave][lr * SP + kc + 16];
    pa[1].h[0] = *(const bf16x8*)&Ps[wave][lr * SP + 32 + kc];
    pa[1].h[1] = *(const bf16x8*)&Ps[wave][lr * SP + 32 + kc + 16];

    // O += P @ V  (B fragments from transposed V tile, rows = d)
#pragma unroll
    for (int j = 0; j < 4; ++j) {
      Frag16 b0, b1;
      int dr = j * 16 + lr;
      b0.h[0] = *(const bf16x8*)&Vc[dr * SP + kc];
      b0.h[1] = *(const bf16x8*)&Vc[dr * SP + kc + 16];
      b1.h[0] = *(const bf16x8*)&Vc[dr * SP + 32 + kc];
      b1.h[1] = *(const bf16x8*)&Vc[dr * SP + 32 + kc + 16];
      o[j] = wmma_bf16(pa[0], b0, o[j]);
      o[j] = wmma_bf16(pa[1], b1, o[j]);
    }
  }

  // normalize + store att[b][t][h*64+d] (bf16)
  const int b = bh >> 4, h = bh & 15;
#pragma unroll
  for (int j = 0; j < 4; ++j)
#pragma unroll
    for (int r = 0; r < 8; ++r) {
      int t = q0 + wave * 16 + r + hi * 8;
      float val = o[j][r] / lsum[r];
      att[(size_t)(b * SEQ + t) * CDIM + h * DH + j * 16 + lr] = (bf16)val;
    }
}

// =====================================================================
// QKV = x @ W_attn   (M=8192, N=3072, K=1024), bf16 WMMA.
// Scatters to Q (scaled by D^-0.5), K as [b,h,t,d], V transposed [b,h,d,t].
// =====================================================================
__global__ __launch_bounds__(256) void qkv_gemm_kernel(
    const float* __restrict__ x, const float* __restrict__ Wa,
    bf16* __restrict__ qb, bf16* __restrict__ kb, bf16* __restrict__ vb)
{
  constexpr int LDA = 40;                 // padded row stride (bf16 elems)
  __shared__ alignas(16) bf16 As[128 * LDA];  // A tile [m][k]
  __shared__ alignas(16) bf16 Bs[128 * LDA];  // B tile transposed [n][k]
  const int tid  = threadIdx.x;
  const int wave = tid >> 5, lane = tid & 31;
  const int wm = wave >> 2, wn = wave & 3;    // 2 x 4 wave grid
  const int lr = lane & 15, hi = lane >> 4;
  const int m0 = blockIdx.x * 128;
  const int n0 = blockIdx.y * 128;

  const v8f zero = {0.f,0.f,0.f,0.f,0.f,0.f,0.f,0.f};
  v8f acc[4][2];
  for (int f = 0; f < 4; ++f) for (int g = 0; g < 2; ++g) acc[f][g] = zero;

  for (int k0 = 0; k0 < CDIM; k0 += 32) {
    __syncthreads();
    // prefetch next K-slab while we stage/compute this one
    if (k0 + 32 < CDIM) {
      __builtin_prefetch(&x[(size_t)(m0 + (tid >> 3)) * CDIM + k0 + 32], 0, 0);
      __builtin_prefetch(&Wa[(size_t)(k0 + 32 + (tid & 31)) * N3C + n0], 0, 0);
    }
    // stage A: 128x32 f32 -> bf16
#pragma unroll
    for (int i = 0; i < 4; ++i) {
      int s = tid + i * 256;
      int row = s >> 3, c4 = (s & 7) << 2;
      float4 f = *(const float4*)&x[(size_t)(m0 + row) * CDIM + k0 + c4];
      bf16x4 hv = { (bf16)f.x, (bf16)f.y, (bf16)f.z, (bf16)f.w };
      *(bf16x4*)&As[row * LDA + c4] = hv;
    }
    // stage B transposed: Bs[n][k] = Wa[k0+k][n0+n]
    {
      int kk = tid & 31, nq = tid >> 5;
#pragma unroll
      for (int i = 0; i < 4; ++i) {
        int n = nq * 16 + i * 4;
        float4 f = *(const float4*)&Wa[(size_t)(k0 + kk) * N3C + n0 + n];
        Bs[(n + 0) * LDA + kk] = (bf16)f.x;
        Bs[(n + 1) * LDA + kk] = (bf16)f.y;
        Bs[(n + 2) * LDA + kk] = (bf16)f.z;
        Bs[(n + 3) * LDA + kk] = (bf16)f.w;
      }
    }
    __syncthreads();

    const int kc = hi * 8;                 // ISA 16-bit A/B lane->K striping
    Frag16 a[4], b[2];
#pragma unroll
    for (int f = 0; f < 4; ++f) {
      int row = wm * 64 + f * 16 + lr;
      a[f].h[0] = *(const bf16x8*)&As[row * LDA + kc];
      a[f].h[1] = *(const bf16x8*)&As[row * LDA + kc + 16];
    }
#pragma unroll
    for (int g = 0; g < 2; ++g) {
      int nr = wn * 32 + g * 16 + lr;
      b[g].h[0] = *(const bf16x8*)&Bs[nr * LDA + kc];
      b[g].h[1] = *(const bf16x8*)&Bs[nr * LDA + kc + 16];
    }
#pragma unroll
    for (int f = 0; f < 4; ++f)
#pragma unroll
      for (int g = 0; g < 2; ++g)
        acc[f][g] = wmma_bf16(a[f], b[g], acc[f][g]);
  }

  // epilogue: scatter to Q (pre-scaled), K, V^T as bf16
#pragma unroll
  for (int f = 0; f < 4; ++f) {
#pragma unroll
    for (int g = 0; g < 2; ++g) {
      int ng  = n0 + wn * 32 + g * 16 + lr;
      int sel = ng >> 10;                // 0=q 1=k 2=v
      int cin = ng & 1023;
      int h = cin >> 6, d = cin & 63;
#pragma unroll
      for (int r = 0; r < 8; ++r) {
        int m = m0 + wm * 64 + f * 16 + r + hi * 8;
        int bidx = m >> 11, t = m & (SEQ - 1);
        float val = acc[f][g][r];
        size_t head = (size_t)(bidx * NH + h);
        if (sel == 0)      qb[(head * SEQ + t) * DH + d] = (bf16)(val * 0.125f);
        else if (sel == 1) kb[(head * SEQ + t) * DH + d] = (bf16)val;
        else               vb[(head * DH + d) * SEQ + t] = (bf16)val;
      }
    }
  }
}

// =====================================================================
// out = att @ W_proj   (M=8192, N=1024, K=1024), f32 output.
// A tile (bf16) staged by TDM while VALU converts the B tile f32->bf16.
// =====================================================================
__global__ __launch_bounds__(256) void proj_gemm_kernel(
    const bf16* __restrict__ att, const float* __restrict__ Wp,
    float* __restrict__ out)
{
  constexpr int LDA = 40;
  __shared__ alignas(16) bf16 As[128 * LDA];
  __shared__ alignas(16) bf16 Bs[128 * LDA];
  const int tid  = threadIdx.x;
  const int wave = tid >> 5, lane = tid & 31;
  const int wm = wave >> 2, wn = wave & 3;
  const int lr = lane & 15, hi = lane >> 4;
  const int m0 = blockIdx.x * 128;
  const int n0 = blockIdx.y * 128;

  const v8f zero = {0.f,0.f,0.f,0.f,0.f,0.f,0.f,0.f};
  v8f acc[4][2];
  for (int f = 0; f < 4; ++f) for (int g = 0; g < 2; ++g) acc[f][g] = zero;

  for (int k0 = 0; k0 < CDIM; k0 += 32) {
    __syncthreads();
#if USE_TDM
    if (wave == 0) {
      // A tile 128 rows x 32 bf16; LDS rows padded to 40 elems
      // (pad 4 DWORDs every 16 DWORDs: interval code 3, amount code 3)
      tdm_load_2d_bf16((unsigned int)(uintptr_t)As,
                       &att[(size_t)m0 * CDIM + k0],
                       /*tile_w=*/32, /*tile_h=*/128,
                       /*tensor_w=*/CDIM, /*tensor_h=*/BQ * SEQ,
                       /*row_stride=*/CDIM, 3u, 3u);
    }
#else
    {   // A tile: already bf16, straight 16B copies
      int row = tid >> 1;
#pragma unroll
      for (int i = 0; i < 2; ++i) {
        int c = (tid & 1) * 16 + i * 8;
        *(bf16x8*)&As[row * LDA + c] =
            *(const bf16x8*)&att[(size_t)(m0 + row) * CDIM + k0 + c];
      }
    }
#endif
    if (k0 + 32 < CDIM)
      __builtin_prefetch(&Wp[(size_t)(k0 + 32 + (tid & 31)) * CDIM + n0], 0, 0);
    {   // B tile transposed, f32 -> bf16
      int kk = tid & 31, nq = tid >> 5;
#pragma unroll
      for (int i = 0; i < 4; ++i) {
        int n = nq * 16 + i * 4;
        float4 f = *(const float4*)&Wp[(size_t)(k0 + kk) * CDIM + n0 + n];
        Bs[(n + 0) * LDA + kk] = (bf16)f.x;
        Bs[(n + 1) * LDA + kk] = (bf16)f.y;
        Bs[(n + 2) * LDA + kk] = (bf16)f.z;
        Bs[(n + 3) * LDA + kk] = (bf16)f.w;
      }
    }
#if USE_TDM
    if (wave == 0) __builtin_amdgcn_s_wait_tensorcnt(0);
#endif
    __syncthreads();

    const int kc = hi * 8;
    Frag16 a[4], b[2];
#pragma unroll
    for (int f = 0; f < 4; ++f) {
      int row = wm * 64 + f * 16 + lr;
      a[f].h[0] = *(const bf16x8*)&As[row * LDA + kc];
      a[f].h[1] = *(const bf16x8*)&As[row * LDA + kc + 16];
    }
#pragma unroll
    for (int g = 0; g < 2; ++g) {
      int nr = wn * 32 + g * 16 + lr;
      b[g].h[0] = *(const bf16x8*)&Bs[nr * LDA + kc];
      b[g].h[1] = *(const bf16x8*)&Bs[nr * LDA + kc + 16];
    }
#pragma unroll
    for (int f = 0; f < 4; ++f)
#pragma unroll
      for (int g = 0; g < 2; ++g)
        acc[f][g] = wmma_bf16(a[f], b[g], acc[f][g]);
  }

#pragma unroll
  for (int f = 0; f < 4; ++f)
#pragma unroll
    for (int g = 0; g < 2; ++g) {
      int ng = n0 + wn * 32 + g * 16 + lr;
#pragma unroll
      for (int r = 0; r < 8; ++r) {
        int m = m0 + wm * 64 + f * 16 + r + hi * 8;
        out[(size_t)m * CDIM + ng] = acc[f][g][r];
      }
    }
}

// =====================================================================
// Host launcher
// =====================================================================
extern "C" void kernel_launch(void* const* d_in, const int* in_sizes, int n_in,
                              void* d_out, int out_size, void* d_ws, size_t ws_size,
                              hipStream_t stream) {
  (void)in_sizes; (void)n_in; (void)out_size; (void)ws_size;
  const float* x  = (const float*)d_in[0];
  const float* Wa = (const float*)d_in[1];
  const float* Wp = (const float*)d_in[2];
  float* out = (float*)d_out;

  const size_t HE = (size_t)BQ * NH * SEQ * DH;   // 8,388,608 elems (16MB bf16)
  bf16* qbuf = (bf16*)d_ws;
  bf16* kbuf = qbuf + HE;
  bf16* vbuf = kbuf + HE;
  bf16* attb = vbuf + HE;                         // total ws: 64MB

  // QKV GEMM: 8192/128 x 3072/128 tiles
  qkv_gemm_kernel<<<dim3(64, 24), 256, 0, stream>>>(x, Wa, qbuf, kbuf, vbuf);
  // Flash attention: (B*H) x (T/64) blocks
  flash_attn_kernel<<<dim3(BQ * NH, SEQ / 64), 128, 0, stream>>>(qbuf, kbuf, vbuf, attb);
  // Projection GEMM: 8192/128 x 1024/128 tiles
  proj_gemm_kernel<<<dim3(64, 8), 256, 0, stream>>>(attb, Wp, out);
}
